// Model_42494406426822
// MI455X (gfx1250) — matrix operand order repaired and checked by reading, not compiled, not run on hardware
//
#include <hip/hip_runtime.h>
#include <stdint.h>

typedef __attribute__((ext_vector_type(16))) _Float16 v16h;
typedef __attribute__((ext_vector_type(8)))  float    v8f;

#define BSZ    32
#define LSEQ   512
#define DMODEL 128
#define DSTATE 16
#define DCONV  4
#define DINNER 256
#define DTRANK 8
#define NMARK  4
#define BL     (BSZ * LSEQ)   // 16384 tokens
#define NCHUNK 8
#define CHLEN  (LSEQ / NCHUNK) // 64

// ---------------------------------------------------------------------------
// 1) per-batch mean / std (ENC_IN == 1)
// ---------------------------------------------------------------------------
__global__ void stats_kernel(const float* __restrict__ x_enc,
                             float* __restrict__ mean, float* __restrict__ stdv) {
    int b = blockIdx.x, t = threadIdx.x;          // 256 threads
    __shared__ float ssum[256], ssq[256];
    float v0 = x_enc[b * LSEQ + t];
    float v1 = x_enc[b * LSEQ + t + 256];
    ssum[t] = v0 + v1;
    ssq[t]  = v0 * v0 + v1 * v1;
    __syncthreads();
    for (int s = 128; s > 0; s >>= 1) {
        if (t < s) { ssum[t] += ssum[t + s]; ssq[t] += ssq[t + s]; }
        __syncthreads();
    }
    if (t == 0) {
        float m   = ssum[0] / (float)LSEQ;
        float var = ssq[0] / (float)LSEQ - m * m;
        mean[b] = m;
        stdv[b] = sqrtf(var + 1e-5f);
    }
}

// ---------------------------------------------------------------------------
// 2) fused token-conv(3, wrap) + temporal embed + positional embed -> x (f16)
// ---------------------------------------------------------------------------
__global__ void embed_kernel(const float* __restrict__ x_enc,
                             const float* __restrict__ x_mark,
                             const float* __restrict__ tok_w,   // [128][1][3]
                             const float* __restrict__ temp_w,  // [4][128]
                             const float* __restrict__ mean,
                             const float* __restrict__ stdv,
                             _Float16* __restrict__ xh) {       // [BL][128] f16
    int bl = blockIdx.x, b = bl >> 9, l = bl & 511;
    int o = threadIdx.x;                          // 128 threads
    __shared__ float xs[3];
    if (o < 3) {
        int li = (l + LSEQ - 1 + o) % LSEQ;       // wrap padding
        xs[o] = (x_enc[b * LSEQ + li] - mean[b]) / stdv[b];
    }
    __syncthreads();
    float tok = xs[0] * tok_w[o * 3 + 0] + xs[1] * tok_w[o * 3 + 1] + xs[2] * tok_w[o * 3 + 2];
    float tmp = 0.f;
#pragma unroll
    for (int m = 0; m < NMARK; ++m)
        tmp += x_mark[bl * NMARK + m] * temp_w[m * DMODEL + o];
    const float cpe = -0.0719738794f;             // -ln(10000)/128
    float div = __expf((float)(o & ~1) * cpe);
    float ang = (float)l * div;
    float pe  = (o & 1) ? __cosf(ang) : __sinf(ang);
    xh[(size_t)bl * DMODEL + o] = (_Float16)(tok + tmp + pe);
}

// ---------------------------------------------------------------------------
// 3) transpose-convert f32 [K][N] weight -> f16 [N][K] (WMMA B operand)
// ---------------------------------------------------------------------------
__global__ void convT_kernel(const float* __restrict__ w, _Float16* __restrict__ wt,
                             int K, int N) {
    int idx = blockIdx.x * blockDim.x + threadIdx.x;
    if (idx >= K * N) return;
    int k = idx / N, n = idx % N;
    wt[(size_t)n * K + k] = (_Float16)w[(size_t)k * N + n];
}

// ---------------------------------------------------------------------------
// 4) WMMA GEMM: C[M,N] = A[M,K](f16 row-major) * Bt[N,K](f16 row-major).
//    One wave -> 16x64 output (A fragment reused over 4 N-tiles, 4 f32
//    accumulators). K templated -> full unroll so the scheduler can pipeline
//    global_load_b128 clauses against v_wmma_f32_16x16x32_f16.
//    Fragment layouts per CDNA5 ISA 7.12.2 (16-bit A 16x32 / B 32x16).
// ---------------------------------------------------------------------------
template<int K>
__global__ void __launch_bounds__(256)
wmma_gemm(const _Float16* __restrict__ A, const _Float16* __restrict__ Bt,
          float* __restrict__ C, int M, int N) {
    int lane  = threadIdx.x;                              // 0..31
    int tileM = blockIdx.y * 8 + threadIdx.y;             // 8 waves / block
    int n4    = blockIdx.x;                               // group of 4 N-tiles
    bool hi   = lane >= 16;
    int lrow  = lane & 15;
    const _Float16* arow = A  + (size_t)(tileM * 16 + lrow) * K;
    const _Float16* b0   = Bt + (size_t)(n4 * 64 + lrow) * K;
    v8f acc0 = {}, acc1 = {}, acc2 = {}, acc3 = {};
#pragma unroll
    for (int k0 = 0; k0 < K; k0 += 32) {
        union { v16h v; uint32_t u[8]; } af, bf0, bf1, bf2, bf3;
#pragma unroll
        for (int v = 0; v < 8; ++v) {
            // A 16x32 f16: lanes0-15 {v0..3:K0..7, v4..7:K16..23}; lanes16-31 +8
            int ka = ((v < 4) ? 0 : 16) + (hi ? 8 : 0) + 2 * (v & 3);
            af.u[v]  = *(const uint32_t*)(arow + k0 + ka);
            // B 32x16 f16: same K order as A: lanes0-15 {v0..3:K0..7, v4..7:K16..23}; lanes16-31 +8
            int kb = ((v < 4) ? 0 : 16) + (hi ? 8 : 0) + 2 * (v & 3);
            bf0.u[v] = *(const uint32_t*)(b0 + (size_t)0 * 16 * K + k0 + kb);
            bf1.u[v] = *(const uint32_t*)(b0 + (size_t)1 * 16 * K + k0 + kb);
            bf2.u[v] = *(const uint32_t*)(b0 + (size_t)2 * 16 * K + k0 + kb);
            bf3.u[v] = *(const uint32_t*)(b0 + (size_t)3 * 16 * K + k0 + kb);
        }
        acc0 = __builtin_amdgcn_wmma_f32_16x16x32_f16(false, af.v, false, bf0.v,
                                                      (short)0, acc0, false, false);
        acc1 = __builtin_amdgcn_wmma_f32_16x16x32_f16(false, af.v, false, bf1.v,
                                                      (short)0, acc1, false, false);
        acc2 = __builtin_amdgcn_wmma_f32_16x16x32_f16(false, af.v, false, bf2.v,
                                                      (short)0, acc2, false, false);
        acc3 = __builtin_amdgcn_wmma_f32_16x16x32_f16(false, af.v, false, bf3.v,
                                                      (short)0, acc3, false, false);
    }
    int nbase = n4 * 64 + lrow;
#pragma unroll
    for (int r = 0; r < 8; ++r) {                         // C/D: vgpr r -> M=r(+8)
        int m = tileM * 16 + r + (hi ? 8 : 0);
        float* crow = C + (size_t)m * N + nbase;
        crow[0]  = acc0[r];
        crow[16] = acc1[r];
        crow[32] = acc2[r];
        crow[48] = acc3[r];
    }
}

// ---------------------------------------------------------------------------
// 5) causal depthwise conv (k=4, zero left-pad) + SiLU on xz[..., :256]
// ---------------------------------------------------------------------------
__global__ void conv_silu_kernel(const float* __restrict__ xz,
                                 const float* __restrict__ conv_w,
                                 const float* __restrict__ conv_b,
                                 float* __restrict__ xi) {
    int bl = blockIdx.x, l = bl & 511, d = threadIdx.x;   // 256 threads
    float acc = conv_b[d];
#pragma unroll
    for (int k = 0; k < DCONV; ++k) {
        int ls = l - (DCONV - 1) + k;
        if (ls >= 0)
            acc += xz[(size_t)(bl - (DCONV - 1) + k) * (2 * DINNER) + d] * conv_w[d * DCONV + k];
    }
    xi[(size_t)bl * DINNER + d] = acc / (1.f + __expf(-acc));
}

// ---------------------------------------------------------------------------
// 6) fused x_proj (256->40) + dt head (8->256, softplus) + B/C extraction
// ---------------------------------------------------------------------------
__global__ void xproj_dt_kernel(const float* __restrict__ xi,
                                const float* __restrict__ x_proj_w, // [256][40]
                                const float* __restrict__ dt_w,     // [8][256]
                                const float* __restrict__ dt_b,
                                float* __restrict__ dt,
                                float* __restrict__ bc) {           // [BL][32]
    int bl = blockIdx.x, t = threadIdx.x;                 // 256 threads
    __shared__ float xs[DINNER];
    __shared__ float dbl[DTRANK + 2 * DSTATE];            // 40
    xs[t] = xi[(size_t)bl * DINNER + t];
    __syncthreads();
    if (t < DTRANK + 2 * DSTATE) {
        float s = 0.f;
        for (int k = 0; k < DINNER; ++k) s += xs[k] * x_proj_w[k * 40 + t];
        dbl[t] = s;
    }
    __syncthreads();
    float r = dt_b[t];
#pragma unroll
    for (int j = 0; j < DTRANK; ++j) r += dbl[j] * dt_w[j * DINNER + t];
    dt[(size_t)bl * DINNER + t] = (r > 20.f) ? r : log1pf(__expf(r));
    if (t < 32) bc[(size_t)bl * 32 + t] = dbl[DTRANK + t];
}

// ---------------------------------------------------------------------------
// 7) chunked parallel selective scan.
//    h_t = a_t h_{t-1} + u_t is associative -> split L=512 into 8 chunks of 64.
//    Block = (8 chunks x 32 channels) for one (batch, channel-group).
//    Pass A: per-chunk product P[n] / offset S[n] -> LDS [chunk][n][lane].
//    Prefix: wave 0 combines 8 chunk-carries per channel.
//    Pass B: replay chunk from carry, fused D-skip + SiLU(z) gate -> f16 out.
//    Critical path 512 -> ~128 sequential steps; 256 blocks instead of 32.
// ---------------------------------------------------------------------------
__global__ void __launch_bounds__(256)
scan_kernel(const float* __restrict__ dt, const float* __restrict__ xi,
            const float* __restrict__ xz, const float* __restrict__ bc,
            const float* __restrict__ A_log, const float* __restrict__ Dp,
            _Float16* __restrict__ yh) {
    int b  = blockIdx.x;                   // batch
    int dg = blockIdx.y;                   // channel group (8 groups of 32)
    int t  = threadIdx.x;                  // 256
    int chunk = t >> 5;                    // wave index = chunk
    int dl    = t & 31;                    // lane = channel in group
    int d     = dg * 32 + dl;

    __shared__ float Ps[NCHUNK][DSTATE][32];
    __shared__ float Ss[NCHUNK][DSTATE][32];
    __shared__ float In[NCHUNK][DSTATE][32];

    float An[DSTATE];
#pragma unroll
    for (int n = 0; n < DSTATE; ++n) An[n] = -__expf(A_log[d * DSTATE + n]);
    float Dv = Dp[d];

    // ---- pass A: chunk-local (P, S) ----
    float h[DSTATE], P[DSTATE];
#pragma unroll
    for (int n = 0; n < DSTATE; ++n) { h[n] = 0.f; P[n] = 1.f; }
    int bl0 = b * LSEQ + chunk * CHLEN;
    for (int s = 0; s < CHLEN; ++s) {
        int bl = bl0 + s;
        float dtv = dt[(size_t)bl * DINNER + d];
        float xiv = xi[(size_t)bl * DINNER + d];
        float dbu = dtv * xiv;
        float Bl[DSTATE];
        const float4* bp = (const float4*)(bc + (size_t)bl * 32);   // wave-uniform
        *(float4*)&Bl[0]  = bp[0]; *(float4*)&Bl[4]  = bp[1];
        *(float4*)&Bl[8]  = bp[2]; *(float4*)&Bl[12] = bp[3];
#pragma unroll
        for (int n = 0; n < DSTATE; ++n) {
            float a = __expf(dtv * An[n]);
            h[n] = a * h[n] + dbu * Bl[n];
            P[n] *= a;
        }
    }
#pragma unroll
    for (int n = 0; n < DSTATE; ++n) {
        Ps[chunk][n][dl] = P[n];
        Ss[chunk][n][dl] = h[n];
    }
    __syncthreads();

    // ---- prefix over chunks (wave 0, one thread per channel) ----
    if (t < 32) {
        float carry[DSTATE];
#pragma unroll
        for (int n = 0; n < DSTATE; ++n) carry[n] = 0.f;
        for (int c = 0; c < NCHUNK; ++c) {
#pragma unroll
            for (int n = 0; n < DSTATE; ++n) {
                In[c][n][dl] = carry[n];
                carry[n] = Ps[c][n][dl] * carry[n] + Ss[c][n][dl];
            }
        }
    }
    __syncthreads();

    // ---- pass B: replay chunk from carry, emit gated output ----
#pragma unroll
    for (int n = 0; n < DSTATE; ++n) h[n] = In[chunk][n][dl];
    for (int s = 0; s < CHLEN; ++s) {
        int bl = bl0 + s;
        float dtv = dt[(size_t)bl * DINNER + d];
        float xiv = xi[(size_t)bl * DINNER + d];
        float dbu = dtv * xiv;
        float BCl[2 * DSTATE];
        const float4* bp = (const float4*)(bc + (size_t)bl * 32);   // wave-uniform
        *(float4*)&BCl[0]  = bp[0]; *(float4*)&BCl[4]  = bp[1];
        *(float4*)&BCl[8]  = bp[2]; *(float4*)&BCl[12] = bp[3];
        *(float4*)&BCl[16] = bp[4]; *(float4*)&BCl[20] = bp[5];
        *(float4*)&BCl[24] = bp[6]; *(float4*)&BCl[28] = bp[7];
        float y = 0.f;
#pragma unroll
        for (int n = 0; n < DSTATE; ++n) {
            float a = __expf(dtv * An[n]);
            h[n] = a * h[n] + dbu * BCl[n];
            y += h[n] * BCl[DSTATE + n];
        }
        y += Dv * xiv;
        float zv = xz[(size_t)bl * (2 * DINNER) + DINNER + d];
        y *= zv / (1.f + __expf(-zv));
        yh[(size_t)bl * DINNER + d] = (_Float16)y;
    }
}

// ---------------------------------------------------------------------------
// 8) head: out_layer (128->1), de-norm, relu, fc0 (1->1), relu
// ---------------------------------------------------------------------------
__global__ void head_kernel(const float* __restrict__ yp,
                            const float* __restrict__ out_layer_w,
                            const float* __restrict__ mean, const float* __restrict__ stdv,
                            const float* __restrict__ fc0_w, const float* __restrict__ fc0_b,
                            float* __restrict__ h1) {
    int bl = blockIdx.x, t = threadIdx.x;                 // 128 threads
    int b = bl >> 9;
    __shared__ float red[128];
    red[t] = yp[(size_t)bl * DMODEL + t] * out_layer_w[t];
    __syncthreads();
    for (int s = 64; s > 0; s >>= 1) {
        if (t < s) red[t] += red[t + s];
        __syncthreads();
    }
    if (t == 0) {
        float xo  = red[0] * stdv[b] + mean[b];
        float dec = fmaxf(xo, 0.f);
        h1[bl] = fmaxf(dec * fc0_w[0] + fc0_b[0], 0.f);
    }
}

// ---------------------------------------------------------------------------
// 9) final MLP: 512 -> 128 -> 64 -> 1 (one block per batch, LDS-resident)
// ---------------------------------------------------------------------------
__global__ void mlp_kernel(const float* __restrict__ h1,
                           const float* __restrict__ fc1_w, const float* __restrict__ fc1_b,
                           const float* __restrict__ fc2_w, const float* __restrict__ fc2_b,
                           const float* __restrict__ fc3_w, const float* __restrict__ fc3_b,
                           float* __restrict__ out) {
    int b = blockIdx.x, t = threadIdx.x;                  // 128 threads
    __shared__ float hrow[LSEQ];
    __shared__ float h2[128];
    __shared__ float h3[64];
    for (int i = t; i < LSEQ; i += 128) hrow[i] = h1[b * LSEQ + i];
    __syncthreads();
    {
        float s = fc1_b[t];
        for (int i = 0; i < LSEQ; ++i) s += hrow[i] * fc1_w[i * 128 + t];
        h2[t] = fmaxf(s, 0.f);
    }
    __syncthreads();
    if (t < 64) {
        float s = fc2_b[t];
        for (int i = 0; i < 128; ++i) s += h2[i] * fc2_w[i * 64 + t];
        h3[t] = fmaxf(s, 0.f);
    }
    __syncthreads();
    if (t == 0) {
        float s = fc3_b[0];
        for (int i = 0; i < 64; ++i) s += h3[i] * fc3_w[i];
        out[b] = s;
    }
}

// ---------------------------------------------------------------------------
extern "C" void kernel_launch(void* const* d_in, const int* in_sizes, int n_in,
                              void* d_out, int out_size, void* d_ws, size_t ws_size,
                              hipStream_t stream) {
    const float* x_enc       = (const float*)d_in[0];
    const float* x_mark_enc  = (const float*)d_in[1];
    const float* tok_w       = (const float*)d_in[4];
    const float* temp_w      = (const float*)d_in[5];
    const float* in_proj_w   = (const float*)d_in[6];
    const float* conv_w      = (const float*)d_in[7];
    const float* conv_b      = (const float*)d_in[8];
    const float* x_proj_w    = (const float*)d_in[9];
    const float* dt_w        = (const float*)d_in[10];
    const float* dt_b        = (const float*)d_in[11];
    const float* A_log       = (const float*)d_in[12];
    const float* Dp          = (const float*)d_in[13];
    const float* out_proj_w  = (const float*)d_in[14];
    const float* out_layer_w = (const float*)d_in[15];
    const float* fc0_w       = (const float*)d_in[16];
    const float* fc0_b       = (const float*)d_in[17];
    const float* fc1_w       = (const float*)d_in[18];
    const float* fc1_b       = (const float*)d_in[19];
    const float* fc2_w       = (const float*)d_in[20];
    const float* fc2_b       = (const float*)d_in[21];
    const float* fc3_w       = (const float*)d_in[22];
    const float* fc3_b       = (const float*)d_in[23];
    (void)in_sizes; (void)n_in; (void)out_size; (void)ws_size;

    char* p = (char*)d_ws;
    auto alloc = [&](size_t bytes) {
        char* r = p;
        p += (bytes + 255) & ~(size_t)255;
        return r;
    };
    float*    mean = (float*)alloc(BSZ * 4);
    float*    stdv = (float*)alloc(BSZ * 4);
    _Float16* xh   = (_Float16*)alloc((size_t)BL * DMODEL * 2);
    _Float16* w1t  = (_Float16*)alloc((size_t)(2 * DINNER) * DMODEL * 2);
    _Float16* w2t  = (_Float16*)alloc((size_t)DMODEL * DINNER * 2);
    float*    xz   = (float*)alloc((size_t)BL * (2 * DINNER) * 4);
    float*    xi   = (float*)alloc((size_t)BL * DINNER * 4);
    float*    dtb  = (float*)alloc((size_t)BL * DINNER * 4);
    float*    bc   = (float*)alloc((size_t)BL * 32 * 4);
    _Float16* yh   = (_Float16*)alloc((size_t)BL * DINNER * 2);
    float*    yp   = (float*)alloc((size_t)BL * DMODEL * 4);
    float*    h1   = (float*)alloc((size_t)BL * 4);

    stats_kernel<<<dim3(BSZ), dim3(256), 0, stream>>>(x_enc, mean, stdv);
    embed_kernel<<<dim3(BL), dim3(DMODEL), 0, stream>>>(x_enc, x_mark_enc, tok_w, temp_w,
                                                        mean, stdv, xh);
    {
        int tot = DMODEL * (2 * DINNER);
        convT_kernel<<<dim3((tot + 255) / 256), dim3(256), 0, stream>>>(in_proj_w, w1t,
                                                                        DMODEL, 2 * DINNER);
    }
    {
        int tot = DINNER * DMODEL;
        convT_kernel<<<dim3((tot + 255) / 256), dim3(256), 0, stream>>>(out_proj_w, w2t,
                                                                        DINNER, DMODEL);
    }
    // in_proj: [16384,128] x [128,512] -> xz   (N-groups of 64, 8 M-tiles/block)
    wmma_gemm<DMODEL><<<dim3((2 * DINNER) / 64, BL / (16 * 8)), dim3(32, 8), 0, stream>>>(
        xh, w1t, xz, BL, 2 * DINNER);
    conv_silu_kernel<<<dim3(BL), dim3(DINNER), 0, stream>>>(xz, conv_w, conv_b, xi);
    xproj_dt_kernel<<<dim3(BL), dim3(DINNER), 0, stream>>>(xi, x_proj_w, dt_w, dt_b, dtb, bc);
    scan_kernel<<<dim3(BSZ, DINNER / 32), dim3(256), 0, stream>>>(dtb, xi, xz, bc,
                                                                  A_log, Dp, yh);
    // out_proj: [16384,256] x [256,128] -> yp
    wmma_gemm<DINNER><<<dim3(DMODEL / 64, BL / (16 * 8)), dim3(32, 8), 0, stream>>>(
        yh, w2t, yp, BL, DMODEL);
    head_kernel<<<dim3(BL), dim3(DMODEL), 0, stream>>>(yp, out_layer_w, mean, stdv,
                                                       fc0_w, fc0_b, h1);
    mlp_kernel<<<dim3(BSZ), dim3(128), 0, stream>>>(h1, fc1_w, fc1_b, fc2_w, fc2_b,
                                                    fc3_w, fc3_b, (float*)d_out);
}
